// SpatialAttention_2740189135324
// MI455X (gfx1250) — matrix operand order repaired and checked by reading, not compiled, not run on hardware
//
#include <hip/hip_runtime.h>

// ---------------------------------------------------------------------------
// SAGAN spatial attention for MI455X (gfx1250, wave32, WMMA bf16 16x16x32).
// B=4, C=256, H=W=64 -> N=4096, DQK=32 (== WMMA K for bf16: perfect fit).
// Round 5: raw v_exp_f32 via __builtin_amdgcn_exp2f (no libm guards), and
// row-max butterfly moved from ds_swizzle to VALU v_permlane16_b32.
// ---------------------------------------------------------------------------

#define BATCH 4
#define CCH   256
#define NPIX  4096
#define DQK   32
#define KC    32      // keys per chunk
#define NW    8       // waves per block (4 query subtiles x 2 channel halves)

typedef __bf16 v16bf __attribute__((ext_vector_type(16)));
typedef __bf16 v8bf  __attribute__((ext_vector_type(8)));
typedef float  v8f   __attribute__((ext_vector_type(8)));

static __device__ inline __bf16 f2bf(float f) {
    union { float f; unsigned u; } v; v.f = f;
    unsigned r = v.u + 0x7fffu + ((v.u >> 16) & 1u);   // round-to-nearest-even
    union { unsigned short s; __bf16 h; } o; o.s = (unsigned short)(r >> 16);
    return o.h;
}

static __device__ inline v16bf cat8(v8bf a, v8bf b) {
    v16bf r;
#pragma unroll
    for (int i = 0; i < 8; ++i) { r[i] = a[i]; r[i + 8] = b[i]; }
    return r;
}

static __device__ inline unsigned lds_off(const void* p) {
    return (unsigned)(unsigned long long)p;   // low 32 bits = LDS offset
}

// packed bf16 max (VOP3P)
static __device__ inline unsigned pkmax(unsigned a, unsigned b) {
    unsigned d;
    asm("v_pk_max_num_bf16 %0, %1, %2" : "=v"(d) : "v"(a), "v"(b));
    return d;
}

// XOR-butterfly within each 16-lane row via v_permlane16_b32 (pure VALU).
static __device__ inline unsigned xlane16(unsigned v, unsigned s0, unsigned s1) {
    return __builtin_amdgcn_permlane16(v, v, s0, s1, false, false);
}

// ---------------------------------------------------------------------------
// Kernel 1: 1x1-conv projections (q,k,v) fp32 -> bf16, WMMA-friendly layouts:
//   qT,kT : [B][N][DQK]   (row = pixel, contiguous over d)
//   vO    : [B][C][N]     (row = channel, contiguous over pixel)
// q is pre-scaled by log2(e) so the attention softmax can use exp2 directly.
// ---------------------------------------------------------------------------
__global__ __launch_bounds__(256) void proj_kernel(
    const float* __restrict__ x,
    const float* __restrict__ qw, const float* __restrict__ qb,
    const float* __restrict__ kw, const float* __restrict__ kb,
    const float* __restrict__ vw, const float* __restrict__ vb,
    __bf16* __restrict__ qT, __bf16* __restrict__ kT, __bf16* __restrict__ vO)
{
    __shared__ float xs[CCH * 64];                 // 64KB x-tile [c][n]

    const int b  = blockIdx.x / (NPIX / 64);
    const int n0 = (blockIdx.x % (NPIX / 64)) * 64;
    const float* xb = x + ((size_t)b * CCH) * NPIX + n0;

#pragma unroll
    for (int i = 0; i < 16; ++i) {
        int idx = threadIdx.x + i * 256;           // 0..4095 float4 slots
        int c   = idx >> 4;
        int nn  = (idx & 15) * 4;
        *(float4*)&xs[c * 64 + nn] = *(const float4*)(xb + (size_t)c * NPIX + nn);
    }
    __syncthreads();

    const int n = threadIdx.x & 63;
    const int g = threadIdx.x >> 6;                // 4 threads share a pixel
    for (int oo = 0; oo < 80; ++oo) {              // [q:32][k:32][v:256]
        int o = g * 80 + oo;
        const float* wrow; float bias;
        if (o < 32)      { wrow = qw + o * CCH;        bias = qb[o]; }
        else if (o < 64) { wrow = kw + (o - 32) * CCH; bias = kb[o - 32]; }
        else             { wrow = vw + (o - 64) * CCH; bias = vb[o - 64]; }
        float acc = bias;
#pragma unroll 8
        for (int c = 0; c < CCH; ++c)
            acc = fmaf(wrow[c], xs[c * 64 + n], acc);
        if (o < 32)        // fold log2(e): softmax done in base-2 domain
            qT[((size_t)b * NPIX + n0 + n) * DQK + o] =
                f2bf(acc * 1.4426950408889634f);
        else if (o < 64)
            kT[((size_t)b * NPIX + n0 + n) * DQK + (o - 32)] = f2bf(acc);
        else
            vO[((size_t)b * CCH + (o - 64)) * NPIX + n0 + n] = f2bf(acc);
    }
}

// ---------------------------------------------------------------------------
// Kernel 2: flash attention. Block = 8 waves, 64 queries, all 256 channels.
// Wave w: query subtile (w>>1)*16, channel half (w&1)*128.
// Keys streamed in 32-wide chunks, double-buffered in LDS via async DMA.
// ---------------------------------------------------------------------------
__global__ __launch_bounds__(256) void attn_kernel(
    const __bf16* __restrict__ qT, const __bf16* __restrict__ kT,
    const __bf16* __restrict__ vO,
    const float* __restrict__ x, const float* __restrict__ gamma,
    float* __restrict__ out)
{
    // LDS layout (44KB), epilogue aliases the front of it:
    //   [0     , 4096 )  Ks : 2 x 32key x 32d   bf16
    //   [4096  , 36864)  Vs : 2 x 256ch x 32key bf16
    //   [36864 , 45056)  Ps : 8 waves x 16q x 32key bf16
    //   epilogue: Osm float[128][65] = 33280 B at offset 0
    __shared__ __align__(64) char smem[45056];
    __bf16* KsB = (__bf16*)smem;
    __bf16* VsB = (__bf16*)(smem + 4096);
    __bf16* PsB = (__bf16*)(smem + 36864);
    float*  Osm = (float*)smem;

    const int tid   = threadIdx.x;
    const int lane  = tid & 31;
    const int wave  = tid >> 5;
    const int qsub  = wave >> 1;
    const int chalf = wave & 1;
    const int b     = blockIdx.x / (NPIX / 64);
    const int nblk  = (blockIdx.x % (NPIX / 64)) * 64;
    const int n0    = nblk + qsub * 16;

    const int l15 = lane & 15;
    const int lhi = lane >> 4;             // 0: lanes 0-15, 1: lanes 16-31

    const __bf16* kbase = kT + (size_t)b * NPIX * DQK;   // uniform -> SGPR pair
    const __bf16* vbase = vO + (size_t)b * CCH * NPIX;   // uniform -> SGPR pair

    // ---- load Q fragment (A layout: lane<16 rows, K0-15; lane>=16 K16-31) ----
    const __bf16* qrow = qT + ((size_t)b * NPIX + n0 + l15) * DQK;
    v8bf qlo = *(const v8bf*)(qrow + lhi * 16);
    v8bf qhi = *(const v8bf*)(qrow + 8 + lhi * 16);
    v16bf qa = cat8(qlo, qhi);

    // ---- "ones" B tile: column 0 all-ones -> rowsum(P) via WMMA ----
    v16bf vones;
    {
        const __bf16 e = (l15 == 0) ? f2bf(1.0f) : f2bf(0.0f);
#pragma unroll
        for (int i = 0; i < 16; ++i) vones[i] = e;
    }

    v8f o[8];
#pragma unroll
    for (int t = 0; t < 8; ++t) o[t] = (v8f)(0.0f);
    v8f osum = (v8f)(0.0f);                // running softmax denominator tile
    float mrow[8];
#pragma unroll
    for (int r = 0; r < 8; ++r) mrow[r] = -1e30f;

    // ---- async DMA of one (K,V) chunk into LDS buffer `dbuf` ----
    auto issue_chunk = [&](int j, int dbuf) {
        unsigned kLds = lds_off(KsB) + (unsigned)(dbuf * 2048 + tid * 8);
        unsigned kOff = (unsigned)(j * DQK + tid * 4) * 2u;
        asm volatile("global_load_async_to_lds_b64 %0, %1, %2"
                     :: "v"(kLds), "v"(kOff), "s"(kbase) : "memory");
        unsigned vLds = lds_off(VsB) + (unsigned)(dbuf * 16384 + tid * 64);
        unsigned vOff = (unsigned)(tid * NPIX + j) * 2u;
#pragma unroll
        for (int i = 0; i < 4; ++i) {
            asm volatile("global_load_async_to_lds_b128 %0, %1, %2"
                         :: "v"(vLds + i * 16u), "v"(vOff + i * 16u), "s"(vbase)
                         : "memory");
        }
    };

    issue_chunk(0, 0);
    asm volatile("s_wait_asynccnt 0" ::: "memory");
    __syncthreads();

    int buf = 0;
    for (int j = 0; j < NPIX; j += KC) {
        const bool more = (j + KC) < NPIX;
        if (more) issue_chunk(j + KC, buf ^ 1);

        // ---- scores: S[16q x 32k] = Q(16x32d) * K^T, 2 WMMAs ----
        v16bf kb0 = *(const v16bf*)&KsB[buf * 1024 + l15 * 32 + lhi * 16];
        v16bf kb1 = *(const v16bf*)&KsB[buf * 1024 + (16 + l15) * 32 + lhi * 16];
        v8f z = (v8f)(0.0f);
        v8f s0 = __builtin_amdgcn_wmma_f32_16x16x32_bf16(
            false, qa, false, kb0, (short)0, z, false, false);
        v8f s1 = __builtin_amdgcn_wmma_f32_16x16x32_bf16(
            false, qa, false, kb1, (short)0, z, false, false);

        // ---- packed-bf16 row-max butterfly on the VALU (permlane16) ----
        float cm[8];
#pragma unroll
        for (int r = 0; r < 8; ++r) cm[r] = fmaxf(s0[r], s1[r]);
        unsigned pk[4];
#pragma unroll
        for (int i = 0; i < 4; ++i)
            pk[i] = (__float_as_uint(cm[2 * i + 1]) & 0xFFFF0000u) |
                    (__float_as_uint(cm[2 * i]) >> 16);
#define RED_STAGE(S0, S1)                                                 \
        {                                                                 \
            unsigned t0 = xlane16(pk[0], S0, S1);                         \
            unsigned t1 = xlane16(pk[1], S0, S1);                         \
            unsigned t2 = xlane16(pk[2], S0, S1);                         \
            unsigned t3 = xlane16(pk[3], S0, S1);                         \
            pk[0] = pkmax(pk[0], t0); pk[1] = pkmax(pk[1], t1);           \
            pk[2] = pkmax(pk[2], t2); pk[3] = pkmax(pk[3], t3);           \
        }
        RED_STAGE(0x67452301u, 0xEFCDAB89u)   // xor 1
        RED_STAGE(0x54761032u, 0xDCFE98BAu)   // xor 2
        RED_STAGE(0x32107654u, 0xBA98FEDCu)   // xor 4
        RED_STAGE(0xFEDCBA98u, 0x76543210u)   // xor 8
#undef RED_STAGE
#pragma unroll
        for (int i = 0; i < 4; ++i) {
            cm[2 * i]     = __uint_as_float(pk[i] << 16);
            cm[2 * i + 1] = __uint_as_float(pk[i] & 0xFFFF0000u);
        }

        // ---- online softmax in base-2 domain (raw v_exp_f32) ----
        float scale[8];
#pragma unroll
        for (int r = 0; r < 8; ++r) {
            float mn = fmaxf(mrow[r], cm[r]);
            scale[r] = __builtin_amdgcn_exp2f(mrow[r] - mn);
            s0[r] = __builtin_amdgcn_exp2f(s0[r] - mn);
            s1[r] = __builtin_amdgcn_exp2f(s1[r] - mn);
            mrow[r] = mn;
        }

        // ---- D-layout -> A-layout via wave-private LDS scratch ----
        {
            const int prow = lhi * 8;
#pragma unroll
            for (int r = 0; r < 8; ++r) {
                PsB[wave * 512 + (prow + r) * 32 + l15]      = f2bf(s0[r]);
                PsB[wave * 512 + (prow + r) * 32 + l15 + 16] = f2bf(s1[r]);
            }
        }
        v8bf plo = *(const v8bf*)&PsB[wave * 512 + l15 * 32 + lhi * 16];
        v8bf phi = *(const v8bf*)&PsB[wave * 512 + l15 * 32 + 8 + lhi * 16];
        v16bf pa = cat8(plo, phi);

        // ---- denominator tile: osum = osum*scale + P x ones (1 WMMA) ----
#pragma unroll
        for (int r = 0; r < 8; ++r) osum[r] *= scale[r];
        osum = __builtin_amdgcn_wmma_f32_16x16x32_bf16(
            false, pa, false, vones, (short)0, osum, false, false);

        // ---- rescale O and accumulate P * V^T over this wave's 128 ch ----
#pragma unroll
        for (int t = 0; t < 8; ++t) {
#pragma unroll
            for (int r = 0; r < 8; ++r) o[t][r] *= scale[r];
            const int c0 = chalf * 128 + t * 16;
            v16bf vb = *(const v16bf*)&VsB[buf * 8192 + (c0 + l15) * 32 + lhi * 16];
            o[t] = __builtin_amdgcn_wmma_f32_16x16x32_bf16(
                false, pa, false, vb, (short)0, o[t], false, false);
        }

        if (more) asm volatile("s_wait_asynccnt 0" ::: "memory");
        __syncthreads();
        buf ^= 1;
    }

    // ---- epilogue: out = gamma*(O/l) + x, coalesced via LDS staging ----
    const float gm = gamma[0];
    float inv[8];
#pragma unroll
    for (int r = 0; r < 8; ++r) {
        float lr = __shfl(osum[r], lhi * 16, 32);   // broadcast column 0
        inv[r] = 1.0f / lr;
    }

    for (int h = 0; h < 2; ++h) {
        if (chalf == h) {
#pragma unroll
            for (int t = 0; t < 8; ++t) {
                const int ch = t * 16 + l15;               // 0..127 within half
#pragma unroll
                for (int r = 0; r < 8; ++r) {
                    const int q = qsub * 16 + lhi * 8 + r; // 0..63
                    Osm[ch * 65 + q] = gm * o[t][r] * inv[r];
                }
            }
        }
        __syncthreads();
        {
            const int n    = tid & 63;
            const int crow = tid >> 6;                     // 0..3
#pragma unroll 4
            for (int i = 0; i < 32; ++i) {
                const int c = crow + i * 4;                // 0..127
                const size_t idx =
                    ((size_t)b * CCH + h * 128 + c) * NPIX + nblk + n;
                out[idx] = Osm[c * 65 + n] + x[idx];       // coalesced ld+st
            }
        }
        __syncthreads();
    }
}

// ---------------------------------------------------------------------------
extern "C" void kernel_launch(void* const* d_in, const int* in_sizes, int n_in,
                              void* d_out, int out_size, void* d_ws, size_t ws_size,
                              hipStream_t stream) {
    const float* x  = (const float*)d_in[0];
    const float* qw = (const float*)d_in[1];
    const float* qb = (const float*)d_in[2];
    const float* kw = (const float*)d_in[3];
    const float* kb = (const float*)d_in[4];
    const float* vw = (const float*)d_in[5];
    const float* vb = (const float*)d_in[6];
    const float* gm = (const float*)d_in[7];
    float* out = (float*)d_out;

    // workspace: qT (1MB) + kT (1MB) + v (8MB), all bf16
    __bf16* qT = (__bf16*)d_ws;
    __bf16* kT = qT + (size_t)BATCH * NPIX * DQK;
    __bf16* vO = kT + (size_t)BATCH * NPIX * DQK;

    proj_kernel<<<BATCH * (NPIX / 64), 256, 0, stream>>>(
        x, qw, qb, kw, kb, vw, vb, qT, kT, vO);
    attn_kernel<<<BATCH * (NPIX / 64), 256, 0, stream>>>(
        qT, kT, vO, x, gm, out);
}
